// Encoder_39694087749856
// MI455X (gfx1250) — matrix operand order repaired and checked by reading, not compiled, not run on hardware
//
#include <hip/hip_runtime.h>

// ---------------------------------------------------------------- types
typedef float  v8f   __attribute__((ext_vector_type(8)));
typedef __bf16 v16bf __attribute__((ext_vector_type(16)));
typedef unsigned int v4u __attribute__((ext_vector_type(4)));

union Frag { v4u q[2]; v16bf v; };   // 32 bytes = 16 bf16 = one WMMA A/B fragment per lane

// ---------------------------------------------------------------- sizes
constexpr int B_    = 64;
constexpr int W_    = 254;
constexpr int S_    = 256;
constexpr int DPHO  = 768;
constexpr int DW_   = 100;
constexpr int DPOS  = 100;
constexpr int H_    = 256;
constexpr int K_IN  = 968;            // D_WORD + D_PHO + D_POS
constexpr int K_PAD = 992;            // 31 * 32
constexpr int KT_IN = K_PAD / 32;     // 31
constexpr int N_TOK = B_ * W_;        // 16256 = 1016 * 16
constexpr int NG    = 2048;           // 2 dirs * 4H columns of fused input projection

// ---------------------------------------------------------------- workspace layout (bytes)
constexpr size_t OFF_CNT = 0;                                   // 2 barrier counters
constexpr size_t OFF_HBF = 256;                                 // h bf16: 2 dir * 2 buf * 64*256
constexpr size_t OFF_EMB = OFF_HBF + 131072;                    // emb bf16 [N_TOK][K_PAD]
constexpr size_t OFF_WIH = OFF_EMB + (size_t)N_TOK * K_PAD * 2; // packed Wih frags
constexpr size_t SZ_WIH  = (size_t)128 * KT_IN * 32 * 8 * 4;    // 128 ntiles * 31 kt * 32 lanes * 8 dw
constexpr size_t OFF_WHH = OFF_WIH + SZ_WIH;
constexpr size_t SZ_WHH  = (size_t)2 * 16 * 4 * 8 * 32 * 8 * 4; // dir,jt,gate,kt,lane,8dw
constexpr size_t OFF_XP  = OFF_WHH + SZ_WHH;                    // xp f32 [N_TOK][2048]
constexpr size_t WS_NEED = OFF_XP + (size_t)N_TOK * NG * 4;

// ---------------------------------------------------------------- helpers
__device__ __forceinline__ unsigned short f2bf(float f) {
  union { float f; unsigned u; } v; v.f = f;
  unsigned r = v.u + 0x7FFFu + ((v.u >> 16) & 1u);   // round-to-nearest-even
  return (unsigned short)(r >> 16);
}

__device__ __forceinline__ float sigm(float x) { return 1.0f / (1.0f + __expf(-x)); }

__device__ __forceinline__ v8f wmma_bf16(v16bf a, v16bf b, v8f c) {
  return __builtin_amdgcn_wmma_f32_16x16x32_bf16(false, a, false, b, (short)0, c, false, false);
}

// 16-bit A-matrix 16x32 fragment from row-major bf16 (contiguous K, matches packed B):
// lanes 0-15 : row = m0+lane,  K = k0+0..15
// lanes 16-31: row = m0+lane-16, K = k0+16..31
__device__ __forceinline__ v16bf load_a_frag(const unsigned short* base, int ld,
                                             int m0, int k0, int lane) {
  int row = m0 + (lane & 15);
  int kk  = k0 + ((lane >> 4) << 4);
  const unsigned short* p = base + (size_t)row * ld + kk;
  Frag f;
  f.q[0] = *(const v4u*)p;
  f.q[1] = *(const v4u*)(p + 8);
  return f.v;
}

// Grid barrier over the 8 workgroups of one direction (monotonic counter).
__device__ __forceinline__ void dir_barrier(unsigned int* cnt, unsigned target) {
  __threadfence();                         // release h writes to device scope
  __syncthreads();
  if (threadIdx.x == 0) {
    __hip_atomic_fetch_add(cnt, 1u, __ATOMIC_RELEASE, __HIP_MEMORY_SCOPE_AGENT);
    while (__hip_atomic_load(cnt, __ATOMIC_ACQUIRE, __HIP_MEMORY_SCOPE_AGENT) < target)
      __builtin_amdgcn_s_sleep(2);
  }
  __syncthreads();
  __threadfence();                         // acquire: invalidate stale near caches
}

// ---------------------------------------------------------------- kernel: zero counters + h buffers
__global__ void k_init(unsigned int* w, int ndw) {
  int i = blockIdx.x * blockDim.x + threadIdx.x;
  if (i < ndw) w[i] = 0u;
}

// ---------------------------------------------------------------- kernel: pack w_ih (both dirs) to WMMA B-fragment order, bf16
// B[k][n] = w[n][k]; fragment layout per lane: lanes 0-15 col n0+lane K k0+0..15,
// lanes 16-31 col n0+lane-16 K k0+16..31, dword j holds (K=2j, K=2j+1).
__global__ void k_pack_wih(const float* __restrict__ wf, const float* __restrict__ wb,
                           unsigned int* __restrict__ packed) {
  int idx = blockIdx.x * 256 + threadIdx.x;     // exactly 128*31*32*8
  int j = idx & 7; int t = idx >> 3;
  int lane = t & 31; t >>= 5;
  int kt = t % KT_IN; int ntile = t / KT_IN;
  int n = ntile * 16 + (lane & 15);
  int k = kt * 32 + ((lane >> 4) << 4) + 2 * j;
  const float* w = (n < 1024) ? (wf + (size_t)n * K_IN) : (wb + (size_t)(n - 1024) * K_IN);
  unsigned lo = (k     < K_IN) ? (unsigned)f2bf(w[k])     : 0u;
  unsigned hi = (k + 1 < K_IN) ? (unsigned)f2bf(w[k + 1]) : 0u;
  packed[idx] = lo | (hi << 16);
}

// ---------------------------------------------------------------- kernel: pack w_hh (both dirs)
__global__ void k_pack_whh(const float* __restrict__ wf, const float* __restrict__ wb,
                           unsigned int* __restrict__ packed) {
  int idx = blockIdx.x * 256 + threadIdx.x;     // exactly 2*16*4*8*32*8
  int j = idx & 7;
  int lane = (idx >> 3) & 31;
  int kt   = (idx >> 8) & 7;
  int g    = (idx >> 11) & 3;
  int jt   = (idx >> 13) & 15;
  int dir  = (idx >> 17) & 1;
  int n = g * 256 + jt * 16 + (lane & 15);      // gate-space column [0,1024)
  int k = kt * 32 + ((lane >> 4) << 4) + 2 * j; // < 256 always
  const float* w = dir ? wb : wf;
  unsigned lo = (unsigned)f2bf(w[(size_t)n * H_ + k]);
  unsigned hi = (unsigned)f2bf(w[(size_t)n * H_ + k + 1]);
  packed[idx] = lo | (hi << 16);
}

// ---------------------------------------------------------------- kernel: build emb (gather + ragged span sum), bf16, K padded to 992
__global__ void k_emb(const int* __restrict__ words, const float* __restrict__ features,
                      const int* __restrict__ lip, const int* __restrict__ postags,
                      const float* __restrict__ wtab, const float* __restrict__ ptab,
                      unsigned short* __restrict__ emb) {
  int n = blockIdx.x;                // token id = b*W + w
  int b = n / W_, w = n % W_;
  int word = words[n];
  int pos  = postags[n];
  int start = lip[b * (W_ + 2) + w];
  int end   = lip[b * (W_ + 2) + w + 1];
  if (start < 0) start = 0; if (start > S_) start = S_;
  if (end   < start) end = start; if (end > S_) end = S_;
  for (int col = threadIdx.x; col < K_PAD; col += blockDim.x) {
    float v = 0.0f;
    if (col < DW_) {
      v = wtab[(size_t)word * DW_ + col];
    } else if (col < DW_ + DPHO) {
      int d = col - DW_;
      float s = 0.0f;
      for (int tk = start; tk < end; ++tk)
        s += features[((size_t)b * S_ + tk) * DPHO + d];
      v = s;
    } else if (col < K_IN) {
      v = ptab[(size_t)pos * DPOS + (col - DW_ - DPHO)];
    }
    emb[(size_t)n * K_PAD + col] = f2bf(v);
  }
}

// ---------------------------------------------------------------- kernel: fused input projection GEMM
// xp[N_TOK][2048] = emb @ [w_ih_f ; w_ih_b]^T + bias. 8 waves/WG, wave = one M-tile,
// B fragments staged once per K-iter in LDS and shared by all waves.
__global__ __launch_bounds__(256) void k_gemm(const unsigned short* __restrict__ emb,
                                              const unsigned int* __restrict__ packedB,
                                              const float* __restrict__ bias_f,
                                              const float* __restrict__ bias_b,
                                              float* __restrict__ xp) {
  __shared__ __align__(16) unsigned int ldsB[8 * 32 * 8];   // 8 N-tiles of B fragments, 8 KB
  int lane = threadIdx.x & 31, wave = threadIdx.x >> 5;
  int mtile = blockIdx.x * 8 + wave;        // 127*8 = 1016 M-tiles
  int ngrp  = blockIdx.y;                   // 16 groups of 128 columns

  v8f acc[8];
  v8f zero = {0.f, 0.f, 0.f, 0.f, 0.f, 0.f, 0.f, 0.f};
#pragma unroll
  for (int i = 0; i < 8; ++i) acc[i] = zero;

  for (int kt = 0; kt < KT_IN; ++kt) {
    __syncthreads();                        // protect previous iteration's reads
    {
      // thread (wave,lane) copies one 32B fragment chunk: frag = wave
      const unsigned int* src =
          packedB + (((size_t)(ngrp * 8 + wave) * KT_IN + kt) * 32 + lane) * 8;
      v4u* dst = (v4u*)&ldsB[(wave * 32 + lane) * 8];
      dst[0] = *(const v4u*)src;
      dst[1] = *(const v4u*)(src + 4);
    }
    __syncthreads();

    v16bf a = load_a_frag(emb, K_PAD, mtile * 16, kt * 32, lane);
#pragma unroll
    for (int nt = 0; nt < 8; ++nt) {
      Frag bf;
      const v4u* p = (const v4u*)&ldsB[(nt * 32 + lane) * 8];
      bf.q[0] = p[0];
      bf.q[1] = p[1];
      acc[nt] = wmma_bf16(a, bf.v, acc[nt]);
    }
  }

  // epilogue: + bias, store f32. C layout: lanes 0-15 rows 0-7, lanes 16-31 rows 8-15.
  int rowb = mtile * 16 + ((lane >> 4) << 3);
  int c0   = lane & 15;
#pragma unroll
  for (int nt = 0; nt < 8; ++nt) {
    int col = ngrp * 128 + nt * 16 + c0;
    float bias = (col < 1024) ? bias_f[col] : bias_b[col - 1024];
#pragma unroll
    for (int r = 0; r < 8; ++r)
      xp[(size_t)(rowb + r) * NG + col] = acc[nt][r] + bias;
  }
}

// ---------------------------------------------------------------- kernel: persistent bidirectional LSTM recurrence
// 16 WGs: blocks 0-7 forward, 8-15 backward. 64 waves per direction = 64 units
// (4 batch M-tiles x 16 column J-tiles). Each wave keeps its 16x16 cell-state
// tile in registers for all 254 steps; h is double-buffered bf16 in global.
__global__ __launch_bounds__(256) void k_lstm(unsigned short* __restrict__ hbf,
                                              const unsigned int* __restrict__ packedWhh,
                                              const float* __restrict__ xp,
                                              float* __restrict__ out,
                                              unsigned int* __restrict__ cnt) {
  int lane = threadIdx.x & 31, wave = threadIdx.x >> 5;
  int dir = blockIdx.x >> 3, wg = blockIdx.x & 7;
  int unit = wg * 8 + wave;                 // 0..63
  int mt = unit >> 4, jt = unit & 15;
  int m0 = mt * 16, j0 = jt * 16;
  int rlocal = (lane >> 4) << 3;            // +8 rows for hi half-wave
  int c0 = lane & 15;

  float creg[8] = {0.f, 0.f, 0.f, 0.f, 0.f, 0.f, 0.f, 0.f};
  float* rnn1 = out;                                              // [B,W,512]
  float* hd   = out + (size_t)N_TOK * 512 + (size_t)dir * N_TOK * 256; // hf then hb
  unsigned int* mycnt = cnt + dir;

  for (int t = 0; t < W_; ++t) {
    int tt = dir ? (W_ - 1 - t) : t;

    // init accumulators from input projection (bias already included)
    v8f acc[4];
#pragma unroll
    for (int g = 0; g < 4; ++g) {
#pragma unroll
      for (int r = 0; r < 8; ++r) {
        int b = m0 + rlocal + r;
        acc[g][r] = xp[(size_t)(b * W_ + tt) * NG + dir * 1024 + g * 256 + j0 + c0];
      }
    }

    // prefetch next step's xp rows (global_prefetch_b8)
    if (t + 1 < W_) {
      int tn = dir ? (W_ - 2 - t) : (t + 1);
      int b = m0 + rlocal;
      __builtin_prefetch(&xp[(size_t)(b * W_ + tn) * NG + dir * 1024 + j0 + c0], 0, 1);
    }

    const unsigned short* hprev = hbf + (size_t)(dir * 2 + (t & 1)) * B_ * H_;
#pragma unroll
    for (int kt = 0; kt < 8; ++kt) {        // K = 256 = 8 * 32
      v16bf a = load_a_frag(hprev, H_, m0, kt * 32, lane);
#pragma unroll
      for (int g = 0; g < 4; ++g) {
        const unsigned int* p =
            packedWhh + ((((size_t)((dir * 16 + jt) * 4 + g)) * 8 + kt) * 32 + lane) * 8;
        Frag bf;
        bf.q[0] = *(const v4u*)p;
        bf.q[1] = *(const v4u*)(p + 4);
        acc[g] = wmma_bf16(a, bf.v, acc[g]);
      }
    }

    // gate nonlinearities: lane-local across the 4 gate tiles (same (row,col))
    unsigned short* hnext = hbf + (size_t)(dir * 2 + ((t + 1) & 1)) * B_ * H_;
#pragma unroll
    for (int r = 0; r < 8; ++r) {
      int b = m0 + rlocal + r;
      int j = j0 + c0;
      float ig = sigm(acc[0][r]);
      float fg = sigm(acc[1][r]);
      float gg = tanhf(acc[2][r]);
      float og = sigm(acc[3][r]);
      float cN = fg * creg[r] + ig * gg;
      creg[r] = cN;
      float h = og * tanhf(cN);
      hnext[(size_t)b * H_ + j] = f2bf(h);
      rnn1[(size_t)(b * W_ + tt) * 512 + dir * 256 + j] = h;
      hd[(size_t)(b * W_ + tt) * 256 + j] = h;
    }

    if (t + 1 < W_) dir_barrier(mycnt, 8u * (unsigned)(t + 1));
  }
}

// ---------------------------------------------------------------- launch
extern "C" void kernel_launch(void* const* d_in, const int* in_sizes, int n_in,
                              void* d_out, int out_size, void* d_ws, size_t ws_size,
                              hipStream_t stream) {
  (void)in_sizes; (void)n_in; (void)out_size;
  if (ws_size < WS_NEED) return;   // workspace too small; nothing safe to do

  const int*   words    = (const int*)d_in[0];
  const float* features = (const float*)d_in[1];
  const int*   lip      = (const int*)d_in[2];
  const int*   postags  = (const int*)d_in[3];
  // d_in[4] chars, d_in[5] masks: unused by reference
  const float* wtab     = (const float*)d_in[6];
  const float* ptab     = (const float*)d_in[7];
  const float* w_ih_f   = (const float*)d_in[8];
  const float* w_hh_f   = (const float*)d_in[9];
  const float* b_f      = (const float*)d_in[10];
  const float* w_ih_b   = (const float*)d_in[11];
  const float* w_hh_b   = (const float*)d_in[12];
  const float* b_b      = (const float*)d_in[13];

  char* ws = (char*)d_ws;
  unsigned int*   cnt  = (unsigned int*)(ws + OFF_CNT);
  unsigned short* hbf  = (unsigned short*)(ws + OFF_HBF);
  unsigned short* emb  = (unsigned short*)(ws + OFF_EMB);
  unsigned int*   pWih = (unsigned int*)(ws + OFF_WIH);
  unsigned int*   pWhh = (unsigned int*)(ws + OFF_WHH);
  float*          xp   = (float*)(ws + OFF_XP);
  float*          out  = (float*)d_out;

  // zero barrier counters + h double-buffers (deterministic per launch)
  k_init<<<129, 256, 0, stream>>>((unsigned int*)ws, (int)(OFF_EMB / 4));
  // pack weights into WMMA B-fragment order (bf16)
  k_pack_wih<<<3968, 256, 0, stream>>>(w_ih_f, w_ih_b, pWih);
  k_pack_whh<<<1024, 256, 0, stream>>>(w_hh_f, w_hh_b, pWhh);
  // build embedding matrix (gather + ragged span pooling), bf16, K padded
  k_emb<<<N_TOK, 256, 0, stream>>>(words, features, lip, postags, wtab, ptab, emb);
  // fused input projection for both directions
  k_gemm<<<dim3(127, 16), 256, 0, stream>>>(emb, pWih, b_f, b_b, xp);
  // persistent bidirectional recurrence
  k_lstm<<<16, 256, 0, stream>>>(hbf, pWhh, xp, out, cnt);
}
